// CrossDimensionalSelfAttention_24154896073397
// MI455X (gfx1250) — matrix operand order repaired and checked by reading, not compiled, not run on hardware
//
#include <hip/hip_runtime.h>
#include <hip/hip_bf16.h>

// ---------------------------------------------------------------------------
// CrossDimensionalSelfAttention for MI455X (gfx1250, wave32, WMMA bf16)
// B=8, H=8, T=1024, D=128.  d_out = [output | attn_time | attn_feature] f32.
// ---------------------------------------------------------------------------

#define T_DIM      1024
#define D_DIM      128
#define HEADS      64                    // B*H
#define HEAD_QKV   (T_DIM * D_DIM)       // 131072 elems per head
#define HEAD_ATTN_T (T_DIM * T_DIM)      // 1048576
#define HEAD_ATTN_F (D_DIM * D_DIM)      // 16384
#define OFF_ATTN_T (HEADS * HEAD_QKV)                    // 8388608
#define OFF_ATTN_F (OFF_ATTN_T + HEADS * HEAD_ATTN_T)    // 75497472

#define INV_TEMP   0.08838834764831845f  // 1/sqrt(128)
#define LPAD       132                   // padded LDS row stride (floats):
                                         // 132*4B = 528B, 16B-aligned, and
                                         // 132 mod 64 banks = 4 -> lanes 0..15
                                         // land in disjoint 4-bank groups.

typedef __attribute__((ext_vector_type(16))) __bf16 bf16x16;
typedef __attribute__((ext_vector_type(8)))  float  f32x8;

union ABReg { bf16x16 v; __bf16 e[16]; };
union CReg  { f32x8  v;  float  e[8];  };

// A-matrix (16x32 bf16) half-index -> K index (contiguous, matches B):
// lanes 0-15: halfs 0..15 -> K0..15
// lanes16-31: halfs 0..15 -> K16..31
__device__ __forceinline__ int a_k(int i, int lane) {
    return i + ((lane >> 4) << 4);
}

__device__ __forceinline__ f32x8 wmma_bf16(bf16x16 a, bf16x16 b, f32x8 c) {
    return __builtin_amdgcn_wmma_f32_16x16x32_bf16(
        /*neg_a=*/false, a, /*neg_b=*/false, b,
        /*c_mod=*/(short)0, c, /*reuse_a=*/false, /*reuse_b=*/false);
}

__device__ __forceinline__ float half_wave_max(float m) {
    #pragma unroll
    for (int off = 1; off < 16; off <<= 1) m = fmaxf(m, __shfl_xor(m, off, 32));
    return m;
}
__device__ __forceinline__ float half_wave_sum(float s) {
    #pragma unroll
    for (int off = 1; off < 16; off <<= 1) s += __shfl_xor(s, off, 32);
    return s;
}

// Fill an A fragment from a row-contiguous f32 source (row stride `stride`).
// Per-lane K pattern is one contiguous 16-float run at +base0,
// so 4 float4 (b128) loads cover the 16 halfs.
__device__ __forceinline__ void load_a_rowmajor(ABReg& a, const float* row,
                                                int lane, float scale) {
    const int base0 = (lane >> 4) << 4;       // 0 or 16
    const float4 q0 = *(const float4*)(row + base0 + 0);
    const float4 q1 = *(const float4*)(row + base0 + 4);
    const float4 q2 = *(const float4*)(row + base0 + 8);
    const float4 q3 = *(const float4*)(row + base0 + 12);
    a.e[ 0] = (__bf16)(q0.x * scale); a.e[ 1] = (__bf16)(q0.y * scale);
    a.e[ 2] = (__bf16)(q0.z * scale); a.e[ 3] = (__bf16)(q0.w * scale);
    a.e[ 4] = (__bf16)(q1.x * scale); a.e[ 5] = (__bf16)(q1.y * scale);
    a.e[ 6] = (__bf16)(q1.z * scale); a.e[ 7] = (__bf16)(q1.w * scale);
    a.e[ 8] = (__bf16)(q2.x * scale); a.e[ 9] = (__bf16)(q2.y * scale);
    a.e[10] = (__bf16)(q2.z * scale); a.e[11] = (__bf16)(q2.w * scale);
    a.e[12] = (__bf16)(q3.x * scale); a.e[13] = (__bf16)(q3.y * scale);
    a.e[14] = (__bf16)(q3.z * scale); a.e[15] = (__bf16)(q3.w * scale);
}

// Fill a B fragment whose per-lane 16 K-values are contiguous in memory.
__device__ __forceinline__ void load_b_contig(ABReg& b, const float* p) {
    const float4 q0 = *(const float4*)(p + 0);
    const float4 q1 = *(const float4*)(p + 4);
    const float4 q2 = *(const float4*)(p + 8);
    const float4 q3 = *(const float4*)(p + 12);
    b.e[ 0] = (__bf16)q0.x; b.e[ 1] = (__bf16)q0.y;
    b.e[ 2] = (__bf16)q0.z; b.e[ 3] = (__bf16)q0.w;
    b.e[ 4] = (__bf16)q1.x; b.e[ 5] = (__bf16)q1.y;
    b.e[ 6] = (__bf16)q1.z; b.e[ 7] = (__bf16)q1.w;
    b.e[ 8] = (__bf16)q2.x; b.e[ 9] = (__bf16)q2.y;
    b.e[10] = (__bf16)q2.z; b.e[11] = (__bf16)q2.w;
    b.e[12] = (__bf16)q3.x; b.e[13] = (__bf16)q3.y;
    b.e[14] = (__bf16)q3.z; b.e[15] = (__bf16)q3.w;
}

// ---------------------------------------------------------------------------
// Kernel 1: attn_feature = softmax_e( (qf/TEMP)^T @ kf )   [128 x 128] / head
// grid = 64 heads, block = 256 (8 waves).  Wave w owns rows d0=16w..16w+15 so
// softmax rows never leave the wave (shfl-only reduction, no LDS reduction).
// qf/kf 32x128 chunks staged in LDS with coalesced b128 loads.
// ---------------------------------------------------------------------------
__global__ void feat_attn(const float* __restrict__ qf,
                          const float* __restrict__ kf,
                          float* __restrict__ attn_f) {
    const int head = blockIdx.x;
    const float* Q  = qf + (size_t)head * HEAD_QKV;
    const float* Kf = kf + (size_t)head * HEAD_QKV;
    float* F = attn_f + (size_t)head * HEAD_ATTN_F;

    const int tid  = threadIdx.x;
    const int lane = tid & 31;
    const int w    = tid >> 5;         // 8 waves
    const int d0   = w * 16;
    const int ln   = lane & 15;
    const int kb   = (lane >> 4) << 4; // B-fragment K base per half-wave

    __shared__ float Qs[32 * LPAD];
    __shared__ float Ks[32 * LPAD];

    CReg acc[8];
    #pragma unroll
    for (int nt = 0; nt < 8; ++nt) acc[nt].v = (f32x8)0.0f;

    for (int t0 = 0; t0 < T_DIM; t0 += 32) {
        __syncthreads();
        // coalesced b128 staging: 1024 float4s, 256 threads -> 4 each
        {
            const float4* Q4 = (const float4*)(Q + t0 * D_DIM);
            const float4* K4 = (const float4*)(Kf + t0 * D_DIM);
            #pragma unroll
            for (int it = 0; it < 4; ++it) {
                const int idx = tid + it * 256;     // [0,1024)
                const int r = idx >> 5, c = idx & 31;
                ((float4*)(Qs + r * LPAD))[c] = Q4[idx];
                ((float4*)(Ks + r * LPAD))[c] = K4[idx];
            }
        }
        __syncthreads();

        ABReg a;                       // A[m=d, k=t] = qf[t, d] / TEMP
        #pragma unroll
        for (int i = 0; i < 16; ++i)
            a.e[i] = (__bf16)(Qs[a_k(i, lane) * LPAD + d0 + ln] * INV_TEMP);
        #pragma unroll
        for (int nt = 0; nt < 8; ++nt) {
            ABReg b;                   // B[k=t, n=e] = kf[t, e]
            #pragma unroll
            for (int i = 0; i < 16; ++i)
                b.e[i] = (__bf16)Ks[(kb + i) * LPAD + nt * 16 + ln];
            acc[nt].v = wmma_bf16(a.v, b.v, acc[nt].v);
        }
    }

    // rowwise softmax; row m = d0 + r + 8*(lane>=16), cols spread over 16 lanes
    #pragma unroll
    for (int r = 0; r < 8; ++r) {
        float m = -__builtin_inff();
        #pragma unroll
        for (int nt = 0; nt < 8; ++nt) m = fmaxf(m, acc[nt].e[r]);
        m = half_wave_max(m);
        float s = 0.0f;
        #pragma unroll
        for (int nt = 0; nt < 8; ++nt) {
            float ev = __expf(acc[nt].e[r] - m);
            acc[nt].e[r] = ev;
            s += ev;
        }
        s = half_wave_sum(s);
        const float rinv = 1.0f / s;
        const int row = d0 + r + ((lane >> 4) << 3);
        #pragma unroll
        for (int nt = 0; nt < 8; ++nt)
            F[row * D_DIM + nt * 16 + ln] = acc[nt].e[r] * rinv;
    }
}

// ---------------------------------------------------------------------------
// Kernel 2: attn_time rows = softmax_s( (q/TEMP) @ k^T )   32 rows per block
// grid = (32 t-blocks, 64 heads), block = 512 (16 waves).
// Wave w: m-tile = w&1, column group n0 = (w>>1)*128 (8 WMMA tiles, K=128).
// The q block (32x128) is staged once in LDS; k^T fragments are per-lane
// contiguous -> b128 global loads.
// ---------------------------------------------------------------------------
__global__ void time_attn(const float* __restrict__ qt,
                          const float* __restrict__ kt,
                          float* __restrict__ attn_t) {
    const int t0   = blockIdx.x * 32;
    const int head = blockIdx.y;
    const float* Q  = qt + (size_t)head * HEAD_QKV;
    const float* Kt = kt + (size_t)head * HEAD_QKV;
    float* P = attn_t + (size_t)head * HEAD_ATTN_T;

    const int tid  = threadIdx.x;
    const int lane = tid & 31;
    const int w    = tid >> 5;           // 16 waves
    const int mt   = w & 1;              // m-tile (rows 16*mt .. +15)
    const int g    = w >> 1;             // column group 0..7
    const int n0   = g * 128;
    const int ln   = lane & 15;
    const int kb   = (lane >> 4) << 4;
    const int rowbase = mt * 16 + ((lane >> 4) << 3); // + r = row in 0..31

    __shared__ float Qs[32 * LPAD];
    __shared__ float red[32][8];
    __shared__ float rowval[32];

    // stage q block once (coalesced b128): 1024 float4s / 512 threads
    {
        const float4* Q4 = (const float4*)(Q + t0 * D_DIM);
        #pragma unroll
        for (int it = 0; it < 2; ++it) {
            const int idx = tid + it * 512;
            const int r = idx >> 5, c = idx & 31;
            ((float4*)(Qs + r * LPAD))[c] = Q4[idx];
        }
    }
    __syncthreads();

    CReg acc[8];
    #pragma unroll
    for (int nt = 0; nt < 8; ++nt) acc[nt].v = (f32x8)0.0f;

    #pragma unroll
    for (int kc = 0; kc < D_DIM; kc += 32) {
        ABReg a;                         // A[m=t, k=d] = q[t0+m, d] / TEMP
        load_a_rowmajor(a, Qs + (mt * 16 + ln) * LPAD + kc, lane, INV_TEMP);
        #pragma unroll
        for (int nt = 0; nt < 8; ++nt) {
            ABReg b;                     // B[k=d, n=s] = k[s, d] (contig in d)
            load_b_contig(b, Kt + (n0 + nt * 16 + ln) * D_DIM + kc + kb);
            acc[nt].v = wmma_bf16(a.v, b.v, acc[nt].v);
        }
    }

    // --- cross-wave softmax over s (1024 cols split across 8 column groups)
    #pragma unroll
    for (int r = 0; r < 8; ++r) {
        float m = -__builtin_inff();
        #pragma unroll
        for (int nt = 0; nt < 8; ++nt) m = fmaxf(m, acc[nt].e[r]);
        m = half_wave_max(m);
        if (ln == 0) red[rowbase + r][g] = m;
    }
    __syncthreads();
    if (tid < 32) {
        float m = -__builtin_inff();
        #pragma unroll
        for (int gg = 0; gg < 8; ++gg) m = fmaxf(m, red[tid][gg]);
        rowval[tid] = m;
    }
    __syncthreads();
    #pragma unroll
    for (int r = 0; r < 8; ++r) {
        const float mx = rowval[rowbase + r];
        float s = 0.0f;
        #pragma unroll
        for (int nt = 0; nt < 8; ++nt) {
            float ev = __expf(acc[nt].e[r] - mx);
            acc[nt].e[r] = ev;
            s += ev;
        }
        s = half_wave_sum(s);
        if (ln == 0) red[rowbase + r][g] = s;
    }
    __syncthreads();
    if (tid < 32) {
        float s = 0.0f;
        #pragma unroll
        for (int gg = 0; gg < 8; ++gg) s += red[tid][gg];
        rowval[tid] = 1.0f / s;
    }
    __syncthreads();
    #pragma unroll
    for (int r = 0; r < 8; ++r) {
        const float rinv = rowval[rowbase + r];
        const int trow = t0 + rowbase + r;
        #pragma unroll
        for (int nt = 0; nt < 8; ++nt)
            P[(size_t)trow * T_DIM + n0 + nt * 16 + ln] = acc[nt].e[r] * rinv;
    }
}

// ---------------------------------------------------------------------------
// Kernel 3: tv = attn_time @ v  (K=1024), then output = tv @ attn_feature.
// grid = (32, 64), block = 512 (16 waves); one 16x16 output tile per wave.
// P rows were just written by this (head, t0) in kernel 2 -> L2-hot readback
// with per-lane b128 loads.  V chunks staged in LDS (coalesced b128).
// ---------------------------------------------------------------------------
__global__ void tv_out(const float* __restrict__ vg,
                       const float* __restrict__ attn_t,
                       const float* __restrict__ attn_f,
                       float* __restrict__ outp) {
    const int t0   = blockIdx.x * 32;
    const int head = blockIdx.y;
    const float* V = vg + (size_t)head * HEAD_QKV;
    const float* P = attn_t + (size_t)head * HEAD_ATTN_T;
    const float* F = attn_f + (size_t)head * HEAD_ATTN_F;
    float* O = outp + (size_t)head * HEAD_QKV;

    const int tid  = threadIdx.x;
    const int lane = tid & 31;
    const int w    = tid >> 5;          // 16 waves
    const int mt   = w & 1;
    const int n0   = (w >> 1) * 16;     // d / e tile
    const int ln   = lane & 15;
    const int kb   = (lane >> 4) << 4;

    __shared__ float Vs[32 * D_DIM];    // 16 KB V chunk staging (row-major)
    __shared__ float TV[32 * LPAD];     // padded tv staging

    const float* Prow = P + (size_t)(t0 + mt * 16 + ln) * T_DIM;

    CReg acc; acc.v = (f32x8)0.0f;
    for (int sc = 0; sc < T_DIM; sc += 32) {
        __syncthreads();
        {   // stage V[sc..sc+31][0..127]: 1024 float4s / 512 threads
            const float4* V4 = (const float4*)(V + sc * D_DIM);
            float4* Vs4 = (float4*)Vs;
            #pragma unroll
            for (int it = 0; it < 2; ++it) {
                const int idx = tid + it * 512;
                Vs4[idx] = V4[idx];
            }
        }
        __syncthreads();

        ABReg a;                        // A[m=t, k=s] = P[t0+m, sc+k]
        load_a_rowmajor(a, Prow + sc, lane, 1.0f);
        ABReg b;                        // B[k=s, n=d] = v[sc+k, d]
        #pragma unroll
        for (int i = 0; i < 16; ++i)
            b.e[i] = (__bf16)Vs[(kb + i) * D_DIM + n0 + ln];
        acc.v = wmma_bf16(a.v, b.v, acc.v);
    }
    #pragma unroll
    for (int r = 0; r < 8; ++r)
        TV[(mt * 16 + r + ((lane >> 4) << 3)) * LPAD + n0 + ln] = acc.e[r];
    __syncthreads();

    CReg acc2; acc2.v = (f32x8)0.0f;
    #pragma unroll
    for (int kc = 0; kc < D_DIM; kc += 32) {
        ABReg a;                        // A[m=t, k=d] = tv[m, d] (b128 LDS)
        load_a_rowmajor(a, TV + (mt * 16 + ln) * LPAD + kc, lane, 1.0f);
        ABReg b;                        // B[k=d, n=e] = attn_feature[d, e]
        #pragma unroll
        for (int i = 0; i < 16; ++i)
            b.e[i] = (__bf16)F[(kc + kb + i) * D_DIM + n0 + ln];
        acc2.v = wmma_bf16(a.v, b.v, acc2.v);
    }
    #pragma unroll
    for (int r = 0; r < 8; ++r)
        O[(size_t)(t0 + mt * 16 + r + ((lane >> 4) << 3)) * D_DIM + n0 + ln] = acc2.e[r];
}

// ---------------------------------------------------------------------------
extern "C" void kernel_launch(void* const* d_in, const int* in_sizes, int n_in,
                              void* d_out, int out_size, void* d_ws, size_t ws_size,
                              hipStream_t stream) {
    const float* qt = (const float*)d_in[0];  // q_time
    const float* kt = (const float*)d_in[1];  // k_time
    const float* qf = (const float*)d_in[2];  // q_feature
    const float* kf = (const float*)d_in[3];  // k_feature
    const float* v  = (const float*)d_in[4];  // v

    float* out    = (float*)d_out;            // [8,8,1024,128]
    float* attn_t = out + OFF_ATTN_T;         // [8,8,1024,1024]
    float* attn_f = out + OFF_ATTN_F;         // [8,8,128,128]

    feat_attn<<<dim3(HEADS), dim3(256), 0, stream>>>(qf, kf, attn_f);
    time_attn<<<dim3(T_DIM / 32, HEADS), dim3(512), 0, stream>>>(qt, kt, attn_t);
    tv_out   <<<dim3(T_DIM / 32, HEADS), dim3(512), 0, stream>>>(v, attn_t, attn_f, out);
}
